// GAT_76613626626157
// MI455X (gfx1250) — matrix operand order repaired and checked by reading, not compiled, not run on hardware
//
#include <hip/hip_runtime.h>
#include <hip/hip_bf16.h>
#include <math.h>

#define NN   50000
#define EE   800000
#define FIN  128
#define HH   4
#define CC   64
#define HC   256
#define GG   64
#define NOUT 5
#define ETOT (EE + NN)   // edges + self loops

typedef __bf16 bf16_t;
typedef bf16_t v16bf __attribute__((ext_vector_type(16)));
typedef bf16_t v4bf  __attribute__((ext_vector_type(4)));
typedef float  v8f   __attribute__((ext_vector_type(8)));

// ---------------------------------------------------------------------------
// helpers
// ---------------------------------------------------------------------------
__device__ __forceinline__ void atomicMaxF(float* addr, float v) {
    // monotone bit-trick float max (works for all sign combos with -inf init)
    if (v >= 0.0f) atomicMax((int*)addr, __float_as_int(v));
    else           atomicMin((unsigned int*)addr, __float_as_uint(v));
}

__device__ __forceinline__ v4bf cvt4(float4 v) {
    v4bf r;
    r[0] = (bf16_t)v.x; r[1] = (bf16_t)v.y;
    r[2] = (bf16_t)v.z; r[3] = (bf16_t)v.w;
    return r;
}

__global__ void k_fill(float* __restrict__ p, float v, int n) {
    int i = blockIdx.x * blockDim.x + threadIdx.x;
    if (i < n) p[i] = v;
}

// ---------------------------------------------------------------------------
// bf16 WMMA GEMM:  C[M,NCOL] = A[M,KDIM] @ W[KDIM,NCOL]   (f32 in, f32 out)
// block = 128 threads (4 waves), block tile 64x64, wave tile 16x64, K step 32
// float4 branch-free staging in, LDS-transposed float4 coalesced stores out
// ---------------------------------------------------------------------------
template<int KDIM, int NCOL>
__global__ __launch_bounds__(128)
void k_gemm_bf16(const float* __restrict__ A, const float* __restrict__ W,
                 float* __restrict__ Cout, int M) {
    __shared__ bf16_t As[64][32];     // A tile, bf16 (4 KB)
    __shared__ bf16_t Ws[32][64];     // W tile, bf16 (4 KB)
    __shared__ float  Cs[64][68];     // C tile, padded (17 KB)

    const int tid  = threadIdx.x;
    const int lane = tid & 31;
    const int wv   = tid >> 5;          // wave id 0..3 -> row sub-tile
    const int m0   = blockIdx.x * 64;
    const int n0   = blockIdx.y * 64;

    v8f acc[4] = {};                    // 4 column tiles of 16

    for (int k0 = 0; k0 < KDIM; k0 += 32) {
        // ---- stage A 64x32 f32 -> bf16 : 512 float4 chunks, 4 per thread
        #pragma unroll
        for (int it = 0; it < 4; ++it) {
            int chunk = it * 128 + tid;
            int r  = chunk >> 3;            // row in tile
            int c4 = (chunk & 7) << 2;      // col (multiple of 4)
            int gr = m0 + r;
            int grc = (gr < M) ? gr : (M - 1);      // clamp: load always valid
            float4 v = *(const float4*)(A + (size_t)grc * KDIM + (k0 + c4));
            if (gr >= M) { v.x = 0.f; v.y = 0.f; v.z = 0.f; v.w = 0.f; }
            *(v4bf*)&As[r][c4] = cvt4(v);
        }
        // ---- stage W 32x64 f32 -> bf16 : 512 float4 chunks, unconditional
        #pragma unroll
        for (int it = 0; it < 4; ++it) {
            int chunk = it * 128 + tid;
            int r  = chunk >> 4;
            int c4 = (chunk & 15) << 2;
            float4 v = *(const float4*)(W + (size_t)(k0 + r) * NCOL + (n0 + c4));
            *(v4bf*)&Ws[r][c4] = cvt4(v);
        }
        __syncthreads();

        // A fragment: 16-bit A 16x32 layout
        //   lanes 0-15 : M=lane,   K = {0..7, 16..23}
        //   lanes16-31 : M=lane-16,K = {8..15,24..31}
        const int am = wv * 16 + (lane & 15);
        const int kb = (lane < 16) ? 0 : 8;
        v16bf afrag;
        #pragma unroll
        for (int i = 0; i < 8; ++i) {
            afrag[i]     = As[am][kb + i];
            afrag[8 + i] = As[am][kb + 16 + i];
        }

        // B fragments: lane l holds N=l%16, K = {kb..kb+7, kb+16..kb+23}
        #pragma unroll
        for (int t = 0; t < 4; ++t) {
            v16bf bfrag;
            #pragma unroll
            for (int i = 0; i < 8; ++i) {
                bfrag[i]     = Ws[kb + i][t * 16 + (lane & 15)];
                bfrag[8 + i] = Ws[kb + 16 + i][t * 16 + (lane & 15)];
            }
            acc[t] = __builtin_amdgcn_wmma_f32_16x16x32_bf16(
                         false, afrag, false, bfrag, (short)0, acc[t],
                         false, false);
        }
        __syncthreads();
    }

    // spill accumulators to LDS C tile
    // C/D layout: VGPR r, lanes 0-15: M=r,N=lane ; lanes 16-31: M=8+r,N=lane-16
    #pragma unroll
    for (int t = 0; t < 4; ++t) {
        #pragma unroll
        for (int r = 0; r < 8; ++r) {
            int row = wv * 16 + ((lane < 16) ? r : r + 8);
            Cs[row][t * 16 + (lane & 15)] = acc[t][r];
        }
    }
    __syncthreads();

    // coalesced float4 stores; full-tile fast path (all but the last block)
    if (m0 + 64 <= M) {
        #pragma unroll
        for (int it = 0; it < 8; ++it) {
            int idx = it * 128 + tid;
            int r = idx >> 4, c4 = (idx & 15) << 2;
            *(float4*)(Cout + (size_t)(m0 + r) * NCOL + (n0 + c4)) =
                *(const float4*)&Cs[r][c4];
        }
    } else {
        #pragma unroll
        for (int it = 0; it < 8; ++it) {
            int idx = it * 128 + tid;
            int r = idx >> 4, c4 = (idx & 15) << 2;
            if (m0 + r < M)
                *(float4*)(Cout + (size_t)(m0 + r) * NCOL + (n0 + c4)) =
                    *(const float4*)&Cs[r][c4];
        }
    }
}

// ---------------------------------------------------------------------------
// per-node attention coefficients: aS[n,h] = sum_c h[n,h,c]*a_s[h,c]
// HEADS/CDIM are powers of two -> all indexing is shifts/masks
// ---------------------------------------------------------------------------
template<int HEADS, int CDIM>
__global__ void k_alpha(const float* __restrict__ h, const float* __restrict__ as_,
                        const float* __restrict__ ad_, float* __restrict__ aS,
                        float* __restrict__ aD) {
    int i = blockIdx.x * blockDim.x + threadIdx.x;
    if (i >= NN * HEADS) return;
    int n = i / HEADS, hh = i & (HEADS - 1);
    const float* hp = h + (size_t)n * (HEADS * CDIM) + hh * CDIM;
    float ss = 0.0f, sd = 0.0f;
    #pragma unroll 4
    for (int c = 0; c < CDIM; c += 4) {
        float4 v = *(const float4*)(hp + c);
        float4 s = *(const float4*)(as_ + hh * CDIM + c);
        float4 d = *(const float4*)(ad_ + hh * CDIM + c);
        ss += v.x * s.x + v.y * s.y + v.z * s.z + v.w * s.w;
        sd += v.x * d.x + v.y * d.y + v.z * d.z + v.w * d.w;
    }
    aS[i] = ss; aD[i] = sd;
}

// ---------------------------------------------------------------------------
// edge pass 1: leaky-relu score + segment max (per dst, per head)
// ---------------------------------------------------------------------------
template<int HEADS>
__global__ void k_edge_score_max(const int* __restrict__ ei,
                                 const float* __restrict__ aS,
                                 const float* __restrict__ aD,
                                 float* __restrict__ sbuf,
                                 float* __restrict__ emax) {
    int i = blockIdx.x * blockDim.x + threadIdx.x;
    if (i >= ETOT * HEADS) return;
    int e = i / HEADS, hh = i & (HEADS - 1);
    int src = (e < EE) ? ei[e]      : (e - EE);
    int dst = (e < EE) ? ei[EE + e] : (e - EE);
    float s = aS[src * HEADS + hh] + aD[dst * HEADS + hh];
    s = (s > 0.0f) ? s : 0.2f * s;          // leaky_relu(0.2)
    sbuf[i] = s;
    atomicMaxF(&emax[dst * HEADS + hh], s);
}

// edge pass 2: exp(s - max) + segment denominator
template<int HEADS>
__global__ void k_edge_expsum(const int* __restrict__ ei,
                              float* __restrict__ sbuf,
                              const float* __restrict__ emax,
                              float* __restrict__ den) {
    int i = blockIdx.x * blockDim.x + threadIdx.x;
    if (i >= ETOT * HEADS) return;
    int e = i / HEADS, hh = i & (HEADS - 1);
    int dst = (e < EE) ? ei[EE + e] : (e - EE);
    float ex = __expf(sbuf[i] - emax[dst * HEADS + hh]);
    sbuf[i] = ex;
    unsafeAtomicAdd(&den[dst * HEADS + hh], ex);    // global_atomic_add_f32
}

// edge pass 3: agg[dst] += (ex/den[dst]) * h[src]
// float4-vectorized gather (L2-resident), native f32 scatter atomics
template<int HEADS, int CDIM>
__global__ void k_edge_agg(const int* __restrict__ ei,
                           const float* __restrict__ h,
                           const float* __restrict__ exb,
                           const float* __restrict__ den,
                           float* __restrict__ agg) {
    constexpr int HC4 = (HEADS * CDIM) >> 2;
    long long idx = (long long)blockIdx.x * blockDim.x + threadIdx.x;
    if (idx >= (long long)ETOT * HC4) return;
    int e = (int)(idx / HC4);                  // power-of-two -> shift
    int t = ((int)idx & (HC4 - 1)) << 2;       // channel (multiple of 4)
    int hh = t / CDIM;                         // power-of-two -> shift
    int src = (e < EE) ? ei[e]      : (e - EE);
    int dst = (e < EE) ? ei[EE + e] : (e - EE);
    float alpha = exb[e * HEADS + hh] / den[dst * HEADS + hh];
    const float4 v = *(const float4*)(h + (size_t)src * (HEADS * CDIM) + t);
    float* ap = agg + (size_t)dst * (HEADS * CDIM) + t;
    unsafeAtomicAdd(ap + 0, alpha * v.x);
    unsafeAtomicAdd(ap + 1, alpha * v.y);
    unsafeAtomicAdd(ap + 2, alpha * v.z);
    unsafeAtomicAdd(ap + 3, alpha * v.w);
}

// ---------------------------------------------------------------------------
// finalize: out = elu(layer_norm(agg + bias)); in-place over agg buffer
// block per node, F threads; fully unrolled reduction trees
// ---------------------------------------------------------------------------
template<int F>
__global__ void k_finalize(float* __restrict__ agg, const float* __restrict__ b,
                           const float* __restrict__ g, const float* __restrict__ be) {
    __shared__ float red[F];
    int n = blockIdx.x, t = threadIdx.x;
    float v = agg[(size_t)n * F + t] + b[t];
    red[t] = v; __syncthreads();
    #pragma unroll
    for (int s = F >> 1; s > 0; s >>= 1) {
        if (t < s) red[t] += red[t + s];
        __syncthreads();
    }
    float mu = red[0] * (1.0f / F); __syncthreads();
    float d = v - mu;
    red[t] = d * d; __syncthreads();
    #pragma unroll
    for (int s = F >> 1; s > 0; s >>= 1) {
        if (t < s) red[t] += red[t + s];
        __syncthreads();
    }
    float var = red[0] * (1.0f / F);
    float y = d * rsqrtf(var + 1e-5f) * g[t] + be[t];
    agg[(size_t)n * F + t] = (y > 0.0f) ? y : (__expf(y) - 1.0f);   // ELU
}

// ---------------------------------------------------------------------------
// global mean pool (segment sums) + tiny MLP head
// ---------------------------------------------------------------------------
__global__ void k_pool(const float* __restrict__ h, const int* __restrict__ batch,
                       float* __restrict__ sums, float* __restrict__ cnt) {
    int i = blockIdx.x * blockDim.x + threadIdx.x;   // over N*16 float4 chunks
    if (i >= NN * (CC / 4)) return;
    int n = i >> 4, c = (i & 15) << 2;
    int g = batch[n];
    const float4 v = *(const float4*)(h + (size_t)n * CC + c);
    float* sp = sums + g * CC + c;
    unsafeAtomicAdd(sp + 0, v.x);
    unsafeAtomicAdd(sp + 1, v.y);
    unsafeAtomicAdd(sp + 2, v.z);
    unsafeAtomicAdd(sp + 3, v.w);
    if (c == 0) unsafeAtomicAdd(&cnt[g], 1.0f);
}

__global__ void k_mlp1(const float* __restrict__ sums, const float* __restrict__ cnt,
                       const float* __restrict__ Wg, const float* __restrict__ bg,
                       float* __restrict__ z1) {
    int g = blockIdx.x, j = threadIdx.x;            // 128 threads
    float invc = 1.0f / fmaxf(cnt[g], 1.0f);
    float s = bg[j];
    #pragma unroll 4
    for (int c = 0; c < CC; ++c) s += (sums[g * CC + c] * invc) * Wg[c * 128 + j];
    z1[g * 128 + j] = fmaxf(s, 0.0f);
}

__global__ void k_mlp2(const float* __restrict__ z1, const float* __restrict__ Wg2,
                       const float* __restrict__ bg2, float* __restrict__ z2) {
    int g = blockIdx.x, j = threadIdx.x;            // 64 threads
    float s = bg2[j];
    #pragma unroll 4
    for (int c = 0; c < 128; ++c) s += z1[g * 128 + c] * Wg2[c * 64 + j];
    z2[g * 64 + j] = fmaxf(s, 0.0f);
}

__global__ void k_mlp3(const float* __restrict__ z2, const float* __restrict__ Wo,
                       const float* __restrict__ bo, float* __restrict__ out) {
    int g = blockIdx.x, j = threadIdx.x;            // 32 threads, 5 valid
    if (j >= NOUT) return;
    float s = bo[j];
    #pragma unroll 4
    for (int c = 0; c < CC; ++c) s += z2[g * CC + c] * Wo[c * NOUT + j];
    out[g * NOUT + j] = s;
}

// ---------------------------------------------------------------------------
extern "C" void kernel_launch(void* const* d_in, const int* in_sizes, int n_in,
                              void* d_out, int out_size, void* d_ws, size_t ws_size,
                              hipStream_t stream) {
    const float* x    = (const float*)d_in[0];
    const int*   ei   = (const int*)  d_in[1];
    const int*   batch= (const int*)  d_in[2];
    const float* W1  = (const float*)d_in[3];
    const float* a1s = (const float*)d_in[4];
    const float* a1d = (const float*)d_in[5];
    const float* b1  = (const float*)d_in[6];
    const float* g1  = (const float*)d_in[7];
    const float* be1 = (const float*)d_in[8];
    const float* W2  = (const float*)d_in[9];
    const float* a2s = (const float*)d_in[10];
    const float* a2d = (const float*)d_in[11];
    const float* b2  = (const float*)d_in[12];
    const float* g2  = (const float*)d_in[13];
    const float* be2 = (const float*)d_in[14];
    const float* W3  = (const float*)d_in[15];
    const float* a3s = (const float*)d_in[16];
    const float* a3d = (const float*)d_in[17];
    const float* b3  = (const float*)d_in[18];
    const float* g3  = (const float*)d_in[19];
    const float* be3 = (const float*)d_in[20];
    const float* Wg  = (const float*)d_in[21];
    const float* bg  = (const float*)d_in[22];
    const float* Wg2 = (const float*)d_in[23];
    const float* bg2 = (const float*)d_in[24];
    const float* Wo  = (const float*)d_in[25];
    const float* bo  = (const float*)d_in[26];
    float* out = (float*)d_out;

    // workspace layout (floats)
    float* ws   = (float*)d_ws;
    float* hA   = ws;                          // N*HC activations / agg (ping)
    float* hLin = hA   + (size_t)NN * HC;      // N*HC gemm output (pong)
    float* aS   = hLin + (size_t)NN * HC;      // N*H
    float* aD   = aS   + (size_t)NN * HH;      // N*H
    float* emax = aD   + (size_t)NN * HH;      // N*H
    float* den  = emax + (size_t)NN * HH;      // N*H
    float* exb  = den  + (size_t)NN * HH;      // ETOT*H
    float* sums = exb  + (size_t)ETOT * HH;    // G*C
    float* cnt  = sums + (size_t)GG * CC;      // G
    float* z1   = cnt  + GG;                   // G*128
    float* z2   = z1   + GG * 128;             // G*64

    const int TB = 256;
    dim3 gemmBlk(128);
    dim3 gemm1Grid((NN + 63) / 64, HC / 64);
    dim3 gemm3Grid((NN + 63) / 64, CC / 64);
    int ehGrid4 = (ETOT * HH + TB - 1) / TB;
    int ehGrid1 = (ETOT + TB - 1) / TB;
    int nhGrid4 = (NN * HH + TB - 1) / TB;
    int nhGrid1 = (NN + TB - 1) / TB;
    int aggGrid4 = (int)(((long long)ETOT * (HC / 4) + TB - 1) / TB);
    int aggGrid1 = (int)(((long long)ETOT * (CC / 4) + TB - 1) / TB);

    // ---------------- layer 1 : [N,128] -> [N,256] ----------------
    k_gemm_bf16<FIN, HC><<<gemm1Grid, gemmBlk, 0, stream>>>(x, W1, hLin, NN);
    k_alpha<HH, CC><<<nhGrid4, TB, 0, stream>>>(hLin, a1s, a1d, aS, aD);
    k_fill<<<nhGrid4, TB, 0, stream>>>(emax, -INFINITY, NN * HH);
    k_fill<<<nhGrid4, TB, 0, stream>>>(den, 0.0f, NN * HH);
    k_fill<<<(NN * HC + TB - 1) / TB, TB, 0, stream>>>(hA, 0.0f, NN * HC);
    k_edge_score_max<HH><<<ehGrid4, TB, 0, stream>>>(ei, aS, aD, exb, emax);
    k_edge_expsum<HH><<<ehGrid4, TB, 0, stream>>>(ei, exb, emax, den);
    k_edge_agg<HH, CC><<<aggGrid4, TB, 0, stream>>>(ei, hLin, exb, den, hA);
    k_finalize<HC><<<NN, HC, 0, stream>>>(hA, b1, g1, be1);

    // ---------------- layer 2 : [N,256] -> [N,256] ----------------
    k_gemm_bf16<HC, HC><<<gemm1Grid, gemmBlk, 0, stream>>>(hA, W2, hLin, NN);
    k_alpha<HH, CC><<<nhGrid4, TB, 0, stream>>>(hLin, a2s, a2d, aS, aD);
    k_fill<<<nhGrid4, TB, 0, stream>>>(emax, -INFINITY, NN * HH);
    k_fill<<<nhGrid4, TB, 0, stream>>>(den, 0.0f, NN * HH);
    k_fill<<<(NN * HC + TB - 1) / TB, TB, 0, stream>>>(hA, 0.0f, NN * HC);
    k_edge_score_max<HH><<<ehGrid4, TB, 0, stream>>>(ei, aS, aD, exb, emax);
    k_edge_expsum<HH><<<ehGrid4, TB, 0, stream>>>(ei, exb, emax, den);
    k_edge_agg<HH, CC><<<aggGrid4, TB, 0, stream>>>(ei, hLin, exb, den, hA);
    k_finalize<HC><<<NN, HC, 0, stream>>>(hA, b2, g2, be2);

    // ---------------- layer 3 : [N,256] -> [N,64], heads=1 --------
    k_gemm_bf16<HC, CC><<<gemm3Grid, gemmBlk, 0, stream>>>(hA, W3, hLin, NN);
    k_alpha<1, CC><<<nhGrid1, TB, 0, stream>>>(hLin, a3s, a3d, aS, aD);
    k_fill<<<nhGrid1, TB, 0, stream>>>(emax, -INFINITY, NN);
    k_fill<<<nhGrid1, TB, 0, stream>>>(den, 0.0f, NN);
    k_fill<<<(NN * CC + TB - 1) / TB, TB, 0, stream>>>(hA, 0.0f, NN * CC);
    k_edge_score_max<1><<<ehGrid1, TB, 0, stream>>>(ei, aS, aD, exb, emax);
    k_edge_expsum<1><<<ehGrid1, TB, 0, stream>>>(ei, exb, emax, den);
    k_edge_agg<1, CC><<<aggGrid1, TB, 0, stream>>>(ei, hLin, exb, den, hA);
    k_finalize<CC><<<NN, CC, 0, stream>>>(hA, b3, g3, be3);

    // ---------------- pool + MLP head -----------------------------
    k_fill<<<(GG * CC + GG + TB - 1) / TB, TB, 0, stream>>>(sums, 0.0f, GG * CC + GG);
    k_pool<<<(NN * (CC / 4) + TB - 1) / TB, TB, 0, stream>>>(hA, batch, sums, cnt);
    k_mlp1<<<GG, 128, 0, stream>>>(sums, cnt, Wg, bg, z1);
    k_mlp2<<<GG, 64, 0, stream>>>(z1, Wg2, bg2, z2);
    k_mlp3<<<GG, 32, 0, stream>>>(z2, Wo, bo, out);
}
